// GNNOptunaModel_21938692948606
// MI455X (gfx1250) — matrix operand order repaired and checked by reading, not compiled, not run on hardware
//
#include <hip/hip_runtime.h>
#include <hip/hip_fp16.h>

// MI455X / gfx1250, wave32. Matrix work done with v_wmma_f32_16x16x32_f16.
//
// NNConv fusion: msg[e,o] = sum_{k,i} h[e,k]*x[src(e),i]*ew2[k,i,o] + x[src]@eb2
//   -> per 16-edge tile: [16 x K] @ [K x 32] GEMM, K = 32*DIN (+1 bias chunk).
// A fragments are synthesized in registers (outer product h (x) x[src]),
// B fragments (rearranged ew2, f16) are staged once per workgroup in LDS in
// the exact per-lane WMMA operand layout (32B/lane conflict-free loads).

typedef __attribute__((ext_vector_type(16))) _Float16 v16h;
typedef __attribute__((ext_vector_type(8)))  float    v8f;

#define WG 256   // 8 waves of 32

// ---------------------------------------------------------------- utilities
__global__ void zero_kernel(float* __restrict__ p, long n) {
  long i = (long)blockIdx.x * blockDim.x + threadIdx.x;
  long st = (long)gridDim.x * blockDim.x;
  for (; i < n; i += st) p[i] = 0.0f;
}

__global__ void f2h_kernel(const float* __restrict__ x, __half* __restrict__ y, int n) {
  int i = blockIdx.x * blockDim.x + threadIdx.x;
  if (i < n) y[i] = __float2half(x[i]);
}

__global__ void cnt_kernel(const int* __restrict__ dst, float* __restrict__ cnt, int E) {
  int e = blockIdx.x * blockDim.x + threadIdx.x;
  if (e < E) atomicAdd(&cnt[dst[e]], 1.0f);
}

// ------------------------------------------------------- fused NNConv edges
template<int DINL>
__launch_bounds__(WG)
__global__ void nnconv_edge(const float* __restrict__ eattr,
                            const int*   __restrict__ esrc,
                            const int*   __restrict__ edst,
                            const float* __restrict__ ew1,
                            const float* __restrict__ eb1,
                            const float* __restrict__ ew2,   // [32, DINL*32]
                            const float* __restrict__ eb2,   // [DINL*32]
                            const __half* __restrict__ xh,   // [N, DINL] f16
                            float* __restrict__ agg,         // [N,32] f32 (zeroed)
                            int E)
{
  constexpr int CH = DINL;                 // k-chunks of 32 over K = 32*DINL
  __shared__ _Float16 bfrag[CH * 2 * 512]; // per-lane B fragments (f16)

  // Build B fragments from ew2 (f32 -> f16) in WMMA B layout:
  //   lane l: col o = tile*16 + (l&15); element t: K = c*32 + (t>>3)*16 + (l>>4)*8 + (t&7)
  for (int idx = threadIdx.x; idx < CH * 2 * 512; idx += WG) {
    int f    = idx >> 9;          // fragment id = c*2 + tile
    int rem  = idx & 511;
    int l    = rem >> 4;          // lane
    int t    = rem & 15;          // element in v16h
    int c    = f >> 1;
    int tile = f & 1;
    int o    = tile * 16 + (l & 15);
    int kk   = c * 32 + (t >> 3) * 16 + (l >> 4) * 8 + (t & 7);
    int kf   = kk / DINL;
    int i    = kk % DINL;
    bfrag[idx] = (_Float16)ew2[(kf * DINL + i) * 32 + o];
  }
  __syncthreads();

  const int lane    = threadIdx.x & 31;
  const int wave    = threadIdx.x >> 5;
  const int half_id = lane >> 4;
  const int o_lo    = lane & 15;

  // Bias-term B fragments (rows = eb2, i.e. the virtual "h=1" feature) in VGPRs.
  v16h bb0{}, bb1{};
  #pragma unroll
  for (int t = 0; t < 16; ++t) {
    int tp = (t >> 3) * 16 + half_id * 8 + (t & 7); // 0..31 within bias chunk
    float v0 = 0.0f, v1 = 0.0f;
    if (!(DINL == 16 && tp >= 16)) {         // DINL==16: upper half is zero-pad
      v0 = eb2[tp * 32 + o_lo];
      v1 = eb2[tp * 32 + 16 + o_lo];
    }
    bb0[t] = (_Float16)v0;
    bb1[t] = (_Float16)v1;
  }

  const int nTiles = (E + 15) >> 4;
  for (int tIdx = blockIdx.x * (WG / 32) + wave; tIdx < nTiles; tIdx += gridDim.x * (WG / 32)) {
    const int  e     = tIdx * 16 + o_lo;     // this lane's edge (dup'd in both halves)
    const bool valid = (e < E);
    const int  ec    = valid ? e : (E - 1);
    const float a    = eattr[ec];
    const int  srcI  = esrc[ec];
    const int  dstI  = edst[ec];
    const _Float16 hs = valid ? (_Float16)1.0f : (_Float16)0.0f;

    // edge MLP hidden: h_k = relu(a*w1_k + b1_k)  (w1/b1 uniform -> s_loads)
    _Float16 hh[32];
    #pragma unroll
    for (int k = 0; k < 32; ++k) {
      float hv = fmaxf(a * ew1[k] + eb1[k], 0.0f);
      hh[k] = hs * (_Float16)hv;
    }

    // gather x[src] row (f16, 16B-vector loads)
    union { uint4 q[DINL / 8]; _Float16 h[DINL]; } xu;
    {
      const uint4* xp = (const uint4*)(xh + (size_t)srcI * DINL);
      #pragma unroll
      for (int qq = 0; qq < DINL / 8; ++qq) xu.q[qq] = xp[qq];
    }
    // per-lane x slices (constant indices + cndmask on half_id)
    _Float16 xsA[8], xsB[8];
    #pragma unroll
    for (int t = 0; t < 8; ++t) {
      xsA[t] = half_id ? xu.h[8 + t] : xu.h[t];
      xsB[t] = (DINL == 32) ? (half_id ? xu.h[24 + t] : xu.h[16 + t]) : (_Float16)0.0f;
    }

    v8f acc0 = {0, 0, 0, 0, 0, 0, 0, 0};
    v8f acc1 = {0, 0, 0, 0, 0, 0, 0, 0};

    #pragma unroll
    for (int c = 0; c < CH; ++c) {
      // A fragment: outer product slice, built in registers.
      v16h av;
      if (DINL == 16) {
        _Float16 h0 = hh[2 * c], h1 = hh[2 * c + 1];
        #pragma unroll
        for (int t = 0; t < 8; ++t) { av[t] = h0 * xsA[t]; av[8 + t] = h1 * xsA[t]; }
      } else {
        _Float16 h0 = hh[c];
        #pragma unroll
        for (int t = 0; t < 8; ++t) { av[t] = h0 * xsA[t]; av[8 + t] = h0 * xsB[t]; }
      }
      v16h b0 = *(const v16h*)&bfrag[(c * 2 + 0) * 512 + lane * 16];
      v16h b1 = *(const v16h*)&bfrag[(c * 2 + 1) * 512 + lane * 16];
      acc0 = __builtin_amdgcn_wmma_f32_16x16x32_f16(false, av, false, b0, (short)0, acc0, false, false);
      acc1 = __builtin_amdgcn_wmma_f32_16x16x32_f16(false, av, false, b1, (short)0, acc1, false, false);
    }
    { // bias chunk: virtual h-feature == 1  ->  adds x[src] @ eb2
      v16h av;
      #pragma unroll
      for (int t = 0; t < 8; ++t) {
        av[t]     = hs * xsA[t];
        av[8 + t] = (DINL == 32) ? (_Float16)(hs * xsB[t]) : (_Float16)0.0f;
      }
      acc0 = __builtin_amdgcn_wmma_f32_16x16x32_f16(false, av, false, bb0, (short)0, acc0, false, false);
      acc1 = __builtin_amdgcn_wmma_f32_16x16x32_f16(false, av, false, bb1, (short)0, acc1, false, false);
    }

    // D layout: lane (half,n) holds rows M = r + 8*half, col n = o_lo.
    #pragma unroll
    for (int r = 0; r < 8; ++r) {
      int M  = r + half_id * 8;
      int d  = __shfl(dstI, M, 32);          // lane M holds edge M's dst
      int ev = tIdx * 16 + M;
      if (ev < E) {
        atomicAdd(&agg[(size_t)d * 32 + o_lo],      acc0[r]);
        atomicAdd(&agg[(size_t)d * 32 + 16 + o_lo], acc1[r]);
      }
    }
  }
}

// -------------------------------------------- node update + BN statistics
template<int DINL>
__launch_bounds__(WG)
__global__ void node_update(const float* __restrict__ xin,   // [N, DINL]
                            const float* __restrict__ root,  // [DINL,32]
                            const float* __restrict__ bias,
                            const float* __restrict__ agg,   // [N,32]
                            const float* __restrict__ cnt,   // [N]
                            float* __restrict__ hout,        // [N,32] pre-BN
                            float* __restrict__ bnsum,       // [64] sum/sumsq
                            int N)
{
  __shared__ float sroot[DINL * 32];
  __shared__ float sred[2][8][32];
  for (int i = threadIdx.x; i < DINL * 32; i += WG) sroot[i] = root[i];
  __syncthreads();
  int o  = threadIdx.x & 31;
  int ln = threadIdx.x >> 5;
  int n  = blockIdx.x * 8 + ln;
  float s = 0.0f;
  if (n < N) {
    s = bias[o] + agg[(size_t)n * 32 + o] / fmaxf(cnt[n], 1.0f);
    #pragma unroll
    for (int i = 0; i < DINL; ++i) s += xin[(size_t)n * DINL + i] * sroot[i * 32 + o];
    hout[(size_t)n * 32 + o] = s;
  }
  sred[0][ln][o] = (n < N) ? s : 0.0f;
  sred[1][ln][o] = (n < N) ? s * s : 0.0f;
  __syncthreads();
  if (ln == 0) {
    float a0 = 0.0f, a1 = 0.0f;
    #pragma unroll
    for (int r = 0; r < 8; ++r) { a0 += sred[0][r][o]; a1 += sred[1][r][o]; }
    atomicAdd(&bnsum[o], a0);
    atomicAdd(&bnsum[32 + o], a1);
  }
}

__global__ void bn_finalize(const float* __restrict__ bnsum,
                            const float* __restrict__ gamma,
                            const float* __restrict__ beta,
                            float* __restrict__ ab, int N)
{
  int o = threadIdx.x;
  if (o < 32) {
    float mu  = bnsum[o] / (float)N;
    float var = bnsum[32 + o] / (float)N - mu * mu;
    float sc  = gamma[o] * rsqrtf(var + 1e-5f);
    ab[o]      = sc;
    ab[32 + o] = beta[o] - mu * sc;
  }
}

__global__ void bn_apply_store(const float* __restrict__ h,
                               const float* __restrict__ ab,
                               float* __restrict__ yf,
                               __half* __restrict__ yh, int total)
{
  int i = blockIdx.x * blockDim.x + threadIdx.x;
  if (i < total) {
    int o = i & 31;
    float v = fmaxf(h[i] * ab[o] + ab[32 + o], 0.0f);
    yf[i] = v;
    yh[i] = __float2half(v);
  }
}

__global__ void bn_apply_pool(const float* __restrict__ h,
                              const float* __restrict__ ab,
                              const int* __restrict__ bids,
                              float* __restrict__ pooled,
                              float* __restrict__ gcnt, int N)
{
  int i = blockIdx.x * blockDim.x + threadIdx.x;
  if (i < N * 32) {
    int o = i & 31;
    int n = i >> 5;
    float v = fmaxf(h[i] * ab[o] + ab[32 + o], 0.0f);
    int g = bids[n];
    atomicAdd(&pooled[g * 32 + o], v);
    if (o == 0) atomicAdd(&gcnt[g], 1.0f);
  }
}

__global__ void readout_kernel(const float* __restrict__ pooled,
                               const float* __restrict__ gcnt,
                               const float* __restrict__ edft,
                               const float* __restrict__ w1,  // [33,64]
                               const float* __restrict__ b1,
                               const float* __restrict__ w2,  // [64]
                               const float* __restrict__ b2,
                               float* __restrict__ out, int G)
{
  int g = blockIdx.x * blockDim.x + threadIdx.x;
  if (g >= G) return;
  float inv = 1.0f / fmaxf(gcnt[g], 1.0f);
  float p[32];
  #pragma unroll
  for (int c = 0; c < 32; ++c) p[c] = pooled[g * 32 + c] * inv;
  float e = edft[g];
  float acc = 0.0f;
  for (int j = 0; j < 64; ++j) {
    float z = b1[j] + e * w1[32 * 64 + j];
    #pragma unroll
    for (int c = 0; c < 32; ++c) z += p[c] * w1[c * 64 + j];
    z = fmaxf(z, 0.0f);
    acc += z * w2[j];
  }
  out[g] = acc + b2[0];
}

// ----------------------------------------------------------------- driver
extern "C" void kernel_launch(void* const* d_in, const int* in_sizes, int n_in,
                              void* d_out, int out_size, void* d_ws, size_t ws_size,
                              hipStream_t stream)
{
  const float* x     = (const float*)d_in[0];
  const float* eattr = (const float*)d_in[1];
  const float* edft  = (const float*)d_in[2];
  const int*   esrc  = (const int*)d_in[3];
  const int*   edst  = (const int*)d_in[4];
  const int*   bids  = (const int*)d_in[5];
  const float* l0_ew1 = (const float*)d_in[6];
  const float* l0_eb1 = (const float*)d_in[7];
  const float* l0_ew2 = (const float*)d_in[8];
  const float* l0_eb2 = (const float*)d_in[9];
  const float* l0_root = (const float*)d_in[10];
  const float* l0_bias = (const float*)d_in[11];
  const float* l0_gamma = (const float*)d_in[12];
  const float* l0_beta  = (const float*)d_in[13];
  const float* l1_ew1 = (const float*)d_in[14];
  const float* l1_eb1 = (const float*)d_in[15];
  const float* l1_ew2 = (const float*)d_in[16];
  const float* l1_eb2 = (const float*)d_in[17];
  const float* l1_root = (const float*)d_in[18];
  const float* l1_bias = (const float*)d_in[19];
  const float* l1_gamma = (const float*)d_in[20];
  const float* l1_beta  = (const float*)d_in[21];
  const float* mlp_w1 = (const float*)d_in[22];
  const float* mlp_b1 = (const float*)d_in[23];
  const float* mlp_w2 = (const float*)d_in[24];
  const float* mlp_b2 = (const float*)d_in[25];
  (void)n_in; (void)out_size; (void)ws_size;

  const int N = in_sizes[0] / 16;
  const int E = in_sizes[1];
  const int G = in_sizes[2];
  const size_t N32 = (size_t)N * 32;

  // workspace layout (accumulated buffers first -> one contiguous zero pass)
  float* w      = (float*)d_ws;
  float* agg0   = w;
  float* agg1   = agg0 + N32;
  float* cnt    = agg1 + N32;               // N
  float* bn0    = cnt + N;                  // 64
  float* bn1    = bn0 + 64;                 // 64
  float* pooled = bn1 + 64;                 // G*32
  float* gcnt   = pooled + (size_t)G * 32;  // G
  long   zeroLen = (long)((gcnt + G) - w);
  float* h0   = gcnt + G;                   // N32
  float* h1   = h0 + N32;                   // N32
  float* x1f  = h1 + N32;                   // N32
  float* ab0  = x1f + N32;                  // 64
  float* ab1  = ab0 + 64;                   // 64
  __half* x0h = (__half*)(ab1 + 64);        // N*16 halfs
  __half* x1h = x0h + (size_t)N * 16;       // N*32 halfs
  float* out  = (float*)d_out;

  zero_kernel<<<1024, WG, 0, stream>>>(w, zeroLen);
  f2h_kernel<<<(N * 16 + WG - 1) / WG, WG, 0, stream>>>(x, x0h, N * 16);
  cnt_kernel<<<(E + WG - 1) / WG, WG, 0, stream>>>(edst, cnt, E);

  // ---- layer 0: NNConv(16 -> 32)
  nnconv_edge<16><<<512, WG, 0, stream>>>(eattr, esrc, edst, l0_ew1, l0_eb1,
                                          l0_ew2, l0_eb2, x0h, agg0, E);
  node_update<16><<<(N + 7) / 8, WG, 0, stream>>>(x, l0_root, l0_bias, agg0, cnt,
                                                  h0, bn0, N);
  bn_finalize<<<1, 32, 0, stream>>>(bn0, l0_gamma, l0_beta, ab0, N);
  bn_apply_store<<<(int)((N32 + WG - 1) / WG), WG, 0, stream>>>(h0, ab0, x1f, x1h, (int)N32);

  // ---- layer 1: NNConv(32 -> 32)
  nnconv_edge<32><<<512, WG, 0, stream>>>(eattr, esrc, edst, l1_ew1, l1_eb1,
                                          l1_ew2, l1_eb2, x1h, agg1, E);
  node_update<32><<<(N + 7) / 8, WG, 0, stream>>>(x1f, l1_root, l1_bias, agg1, cnt,
                                                  h1, bn1, N);
  bn_finalize<<<1, 32, 0, stream>>>(bn1, l1_gamma, l1_beta, ab1, N);
  bn_apply_pool<<<(int)((N32 + WG - 1) / WG), WG, 0, stream>>>(h1, ab1, bids, pooled, gcnt, N);

  // ---- readout MLP
  readout_kernel<<<(G + 63) / 64, 64, 0, stream>>>(pooled, gcnt, edft, mlp_w1,
                                                   mlp_b1, mlp_w2, mlp_b2, out, G);
}
